// EncoderModel_GRU_58952721105033
// MI455X (gfx1250) — compile-verified, matrix operand order checked by reading, not run on hardware
//
#include <hip/hip_runtime.h>
#include <hip/hip_bf16.h>
#include <stdint.h>

// ---------------------------------------------------------------------------
// GRU encoder final state, MI455X (gfx1250, wave32, WMMA bf16).
//   B=128, T=512, I=256, H=512, G=3H=1536
// Phase A: convert X->[T,B,I] bf16, W_ih/W_hh -> bf16
// Phase B: x_gates[T,B,3H] = Xbf @ W_ih^T + b_ih   (v_wmma_f32_16x16x32_bf16)
// Phase C: persistent kernel, 32 blocks x 8 waves; W_hh slice cached in LDS;
//          per step: hg = h @ W_hh^T (WMMA from LDS), gates, global barrier.
// ---------------------------------------------------------------------------

#define BB 128
#define TT 512
#define II 256
#define HH 512
#define GG 1536            // 3*H

typedef __attribute__((ext_vector_type(16))) __bf16 v16bf;
typedef __attribute__((ext_vector_type(8)))  __bf16 v8bf;
typedef __attribute__((ext_vector_type(8)))  float  v8f;

// ---- workspace layout (bytes) ----
static const size_t OFF_XBF  = 0;                                   // T*B*I bf16  = 32 MB
static const size_t OFF_WIH  = OFF_XBF + (size_t)TT*BB*II*2;        // 1536*256 bf16
static const size_t OFF_WHH  = OFF_WIH + (size_t)GG*II*2;           // 1536*512 bf16
static const size_t OFF_XG   = ((OFF_WHH + (size_t)GG*HH*2) + 255) & ~(size_t)255; // T*B*G f32 = 384 MB
static const size_t OFF_HBF  = ((OFF_XG + (size_t)TT*BB*GG*4) + 255) & ~(size_t)255; // 2*B*H bf16
static const size_t OFF_HF   = OFF_HBF + (size_t)2*BB*HH*2;         // 2*B*H f32
static const size_t OFF_CNT  = OFF_HF  + (size_t)2*BB*HH*4;         // int barrier counter

__device__ inline v16bf pack16(v8bf lo, v8bf hi) {
    return __builtin_shufflevector(lo, hi, 0,1,2,3,4,5,6,7,8,9,10,11,12,13,14,15);
}

// ---------------- Phase A kernels ----------------
__global__ void k_convert_weights(const float* __restrict__ wih,
                                  const float* __restrict__ whh,
                                  __bf16* __restrict__ wih_bf,
                                  __bf16* __restrict__ whh_bf) {
    int tid = blockIdx.x * blockDim.x + threadIdx.x;
    const int NIH = GG * II;
    const int NHH = GG * HH;
    if (tid < NIH)             wih_bf[tid]        = (__bf16)wih[tid];
    else if (tid < NIH + NHH)  whh_bf[tid - NIH]  = (__bf16)whh[tid - NIH];
}

__global__ void k_convert_x(const float* __restrict__ x,   // [B,T,I]
                            __bf16* __restrict__ xbf) {    // [T,B,I]
    size_t tid = (size_t)blockIdx.x * blockDim.x + threadIdx.x;   // B*T*I threads
    int b = (int)(tid / ((size_t)TT * II));
    int rem = (int)(tid % ((size_t)TT * II));
    int t = rem / II;
    int i = rem % II;
    xbf[((size_t)t * BB + b) * II + i] = (__bf16)x[tid];
}

__global__ void k_init(__bf16* __restrict__ hbf0, float* __restrict__ hf0,
                       int* __restrict__ cnt) {
    int tid = blockIdx.x * blockDim.x + threadIdx.x;   // 65536 threads
    hbf0[tid] = (__bf16)0.0f;
    hf0[tid]  = 0.0f;
    if (tid == 0) *cnt = 0;
}

// ---------------- Phase B: x_gates GEMM ----------------
// D[r, g] = sum_k Xbf[r,k] * Wih[g,k] + b_ih[g],  r = t*128+b  (4096x96 tiles)
__global__ void __launch_bounds__(256)
k_gemm_xgates(const __bf16* __restrict__ xbf,     // [T*B, I]
              const __bf16* __restrict__ wih_bf,  // [G, I]
              const float*  __restrict__ b_ih,    // [G]
              float* __restrict__ xg) {           // [T*B, G]
    const int lane  = threadIdx.x & 31;
    const int wave  = threadIdx.x >> 5;
    const int tile  = blockIdx.x * 8 + wave;      // 393216 tiles total
    const int mt    = tile / 96;
    const int nt    = tile % 96;
    const int ln    = lane & 15;
    const int khalf = (lane < 16) ? 0 : 8;

    const __bf16* ap = xbf    + (size_t)(mt * 16 + ln) * II;
    const __bf16* bp = wih_bf + (size_t)(nt * 16 + ln) * II;

    v8f acc = {};
#pragma unroll
    for (int kt = 0; kt < II / 32; ++kt) {
        const int k0 = kt * 32 + khalf;
        v16bf a = pack16(*(const v8bf*)(ap + k0), *(const v8bf*)(ap + k0 + 16));
        v16bf b = pack16(*(const v8bf*)(bp + k0), *(const v8bf*)(bp + k0 + 16));
        acc = __builtin_amdgcn_wmma_f32_16x16x32_bf16(false, a, false, b,
                                                      (short)0, acc, false, false);
    }
    const float bias = b_ih[nt * 16 + ln];
    const int rbase = mt * 16 + ((lane < 16) ? 0 : 8);
#pragma unroll
    for (int i = 0; i < 8; ++i) {
        xg[(size_t)(rbase + i) * GG + nt * 16 + ln] = acc[i] + bias;
    }
}

// ---------------- Phase C: persistent recurrence ----------------
__device__ inline void grid_barrier(int* cnt, int target) {
    __syncthreads();
    if (threadIdx.x == 0) {
        __hip_atomic_fetch_add(cnt, 1, __ATOMIC_ACQ_REL, __HIP_MEMORY_SCOPE_AGENT);
        while (__hip_atomic_load(cnt, __ATOMIC_ACQUIRE, __HIP_MEMORY_SCOPE_AGENT) < target) {
            __builtin_amdgcn_s_sleep(2);
        }
    }
    __syncthreads();
}

__global__ void __launch_bounds__(256)
k_gru_recurrence(const __bf16* __restrict__ whh_bf,  // [G, H] row-major
                 const float*  __restrict__ b_hh,    // [G]
                 const float*  __restrict__ xg,      // [T, B, G]
                 __bf16* __restrict__ hbf,           // [2][B*H]
                 float*  __restrict__ hf,            // [2][B*H]
                 float*  __restrict__ out,           // [B*H]
                 int* __restrict__ cnt) {
    const int j     = blockIdx.x;            // column group: H-cols j*16..j*16+15
    const int lane  = threadIdx.x & 31;
    const int mt    = threadIdx.x >> 5;      // row tile 0..7 -> batch rows mt*16..+15
    const int ln    = lane & 15;
    const int khalf = (lane < 16) ? 0 : 8;
    const int col   = j * 16 + ln;           // H column for C/D lanes

    // LDS: W_hh rows for the 48 gate-columns owned by this block, [48][H] bf16.
    __shared__ __bf16 sB[48 * HH];
    {
        // 48 rows x 512 bf16 = 48 rows x 64 uint4
        for (int v = threadIdx.x; v < 48 * (HH / 8); v += 256) {
            int c  = v >> 6;                 // 0..47
            int kk = (v & 63);               // uint4 index in row
            int g  = (c / 16) * HH + j * 16 + (c % 16);
            ((uint4*)sB)[c * (HH / 8) + kk] =
                ((const uint4*)whh_bf)[(size_t)g * (HH / 8) + kk];
        }
    }
    const float bhr = b_hh[0 * HH + col];
    const float bhz = b_hh[1 * HH + col];
    const float bhn = b_hh[2 * HH + col];
    __syncthreads();

    for (int t = 0; t < TT; ++t) {
        const int cur = t & 1, nxt = cur ^ 1;
        const __bf16* ha = hbf + (size_t)cur * BB * HH + (size_t)(mt * 16 + ln) * HH;

        v8f ar = {}, az = {}, an = {};
#pragma unroll 4
        for (int kt = 0; kt < HH / 32; ++kt) {
            const int k0 = kt * 32 + khalf;
            v16bf a = pack16(*(const v8bf*)(ha + k0), *(const v8bf*)(ha + k0 + 16));
            const __bf16* b0 = sB + (size_t)(0 * 16 + ln) * HH + k0;
            const __bf16* b1 = sB + (size_t)(1 * 16 + ln) * HH + k0;
            const __bf16* b2 = sB + (size_t)(2 * 16 + ln) * HH + k0;
            v16bf br = pack16(*(const v8bf*)(b0), *(const v8bf*)(b0 + 16));
            v16bf bz = pack16(*(const v8bf*)(b1), *(const v8bf*)(b1 + 16));
            v16bf bn = pack16(*(const v8bf*)(b2), *(const v8bf*)(b2 + 16));
            ar = __builtin_amdgcn_wmma_f32_16x16x32_bf16(false, a, false, br, (short)0, ar, false, false);
            az = __builtin_amdgcn_wmma_f32_16x16x32_bf16(false, a, false, bz, (short)0, az, false, false);
            an = __builtin_amdgcn_wmma_f32_16x16x32_bf16(false, a, false, bn, (short)0, an, false, false);
        }

        const float* xgs = xg + (size_t)t * BB * GG;
        // prefetch next timestep's x_gates slice into L2
        if (t + 1 < TT) {
            size_t poff = (size_t)(blockIdx.x * 256 + threadIdx.x) * 96;
            if (poff < (size_t)BB * GG * 4)
                __builtin_prefetch((const char*)(xg + (size_t)(t + 1) * BB * GG) + poff, 0, 1);
        }

        const int rbase = mt * 16 + ((lane < 16) ? 0 : 8);
#pragma unroll
        for (int i = 0; i < 8; ++i) {
            const int b = rbase + i;
            const float xr = xgs[(size_t)b * GG + 0 * HH + col];
            const float xz = xgs[(size_t)b * GG + 1 * HH + col];
            const float xn = xgs[(size_t)b * GG + 2 * HH + col];
            const float hp = hf[(size_t)cur * BB * HH + (size_t)b * HH + col];
            const float hr = ar[i] + bhr;
            const float hz = az[i] + bhz;
            const float hn = an[i] + bhn;
            const float r = 1.0f / (1.0f + __expf(-(xr + hr)));
            const float z = 1.0f / (1.0f + __expf(-(xz + hz)));
            const float narg = xn + r * hn;
            const float e = __expf(-2.0f * narg);
            const float n = (1.0f - e) / (1.0f + e);
            const float hnew = (1.0f - z) * n + z * hp;
            hf[(size_t)nxt * BB * HH + (size_t)b * HH + col] = hnew;
            hbf[(size_t)nxt * BB * HH + (size_t)b * HH + col] = (__bf16)hnew;
            if (t == TT - 1) out[(size_t)b * HH + col] = hnew;
        }
        grid_barrier(cnt, 32 * (t + 1));
    }
}

// ---------------- host launch ----------------
extern "C" void kernel_launch(void* const* d_in, const int* in_sizes, int n_in,
                              void* d_out, int out_size, void* d_ws, size_t ws_size,
                              hipStream_t stream) {
    const float* X    = (const float*)d_in[0];   // [B,T,I]
    const float* Wih  = (const float*)d_in[1];   // [3H,I]
    const float* Whh  = (const float*)d_in[2];   // [3H,H]
    const float* bih  = (const float*)d_in[3];   // [3H]
    const float* bhh  = (const float*)d_in[4];   // [3H]
    float* out = (float*)d_out;

    char* ws = (char*)d_ws;
    __bf16* xbf    = (__bf16*)(ws + OFF_XBF);
    __bf16* wih_bf = (__bf16*)(ws + OFF_WIH);
    __bf16* whh_bf = (__bf16*)(ws + OFF_WHH);
    float*  xg     = (float*) (ws + OFF_XG);
    __bf16* hbf    = (__bf16*)(ws + OFF_HBF);
    float*  hf     = (float*) (ws + OFF_HF);
    int*    cnt    = (int*)   (ws + OFF_CNT);

    // Phase A
    {
        int n = GG * II + GG * HH;
        k_convert_weights<<<(n + 255) / 256, 256, 0, stream>>>(Wih, Whh, wih_bf, whh_bf);
    }
    {
        size_t n = (size_t)BB * TT * II;
        k_convert_x<<<(unsigned)(n / 256), 256, 0, stream>>>(X, xbf);
    }
    k_init<<<BB * HH / 256, 256, 0, stream>>>(hbf, hf, cnt);

    // Phase B: 4096 x 96 output tiles, 8 waves/block
    k_gemm_xgates<<<(4096 * 96) / 8, 256, 0, stream>>>(xbf, wih_bf, bih, xg);

    // Phase C: persistent, 32 blocks (all resident), device-wide barrier per step
    k_gru_recurrence<<<32, 256, 0, stream>>>(whh_bf, bhh, xg, hbf, hf, out, cnt);
}
